// NCAWeightGenerator_10677288698393
// MI455X (gfx1250) — matrix operand order repaired and checked by reading, not compiled, not run on hardware
//
#include <hip/hip_runtime.h>
#include <hip/hip_bf16.h>
#include <cstdint>

// ---------------------------------------------------------------------------
// NCA weight-generator for MI455X (gfx1250, wave32).
//   * 64 sequential conv steps -> 64 kernel launches ping-ponging d_ws buffers
//   * conv3x3 (8->32) as implicit GEMM on v_wmma_f32_16x16x32_f16 (K=72 pad 96)
//   * conv1x1 (32->8) as one K=32 WMMA (N padded to 16)
//   * halo tile staged by the Tensor Data Mover (tensor_load_to_lds) on
//     interior tiles; bounds-checked loop on the 252 edge tiles
//   * exact JAX threefry2x32 dropout mask (keys split on host)
// State (32 MB) fits in the 192 MB L2; HBM traffic per step is negligible.
// ---------------------------------------------------------------------------

#define IMG   1024
#define HW    (IMG * IMG)
#define SDIM  8
#define HID   32

#if defined(__HIP_DEVICE_COMPILE__) && __has_builtin(__builtin_amdgcn_tensor_load_to_lds) && __has_builtin(__builtin_amdgcn_s_wait_tensorcnt)
#define USE_TDM 1
#else
#define USE_TDM 0
#endif

typedef __attribute__((ext_vector_type(16))) _Float16 v16h;
typedef __attribute__((ext_vector_type(8)))  float    v8f;
typedef __attribute__((ext_vector_type(4)))  unsigned u32x4;
typedef __attribute__((ext_vector_type(8)))  int      i32x8;
typedef __attribute__((ext_vector_type(4)))  int      i32x4;

union F16x16 { v16h v; _Float16 h[16]; };

__host__ __device__ inline unsigned rotl32(unsigned x, int r) {
  return (x << r) | (x >> (32 - r));
}

// JAX threefry2x32 (20 rounds, key-injection every 4)
__host__ __device__ inline void threefry2x32(unsigned k0, unsigned k1,
                                             unsigned x0, unsigned x1,
                                             unsigned &o0, unsigned &o1) {
  unsigned ks0 = k0, ks1 = k1, ks2 = k0 ^ k1 ^ 0x1BD11BDAu;
  const int R[8] = {13, 15, 26, 6, 17, 29, 16, 24};
  x0 += ks0; x1 += ks1;
  for (int g = 0; g < 5; ++g) {
    const int *r = (g & 1) ? (R + 4) : R;
    for (int i = 0; i < 4; ++i) { x0 += x1; x1 = rotl32(x1, r[i]); x1 ^= x0; }
    switch (g) {
      case 0: x0 += ks1; x1 += ks2 + 1u; break;
      case 1: x0 += ks2; x1 += ks0 + 2u; break;
      case 2: x0 += ks0; x1 += ks1 + 3u; break;
      case 3: x0 += ks1; x1 += ks2 + 4u; break;
      case 4: x0 += ks2; x1 += ks0 + 5u; break;
    }
  }
  o0 = x0; o1 = x1;
}

// mask = (uniform(key, HW)[p] > 0.5) with JAX's split-half counter layout
__device__ inline float nca_mask(unsigned k0, unsigned k1, unsigned p) {
  const unsigned half = (unsigned)(HW / 2);
  unsigned o0, o1, bits;
  if (p < half) { threefry2x32(k0, k1, p, p + half, o0, o1); bits = o0; }
  else          { threefry2x32(k0, k1, p - half, p, o0, o1); bits = o1; }
  float u = __uint_as_float((bits >> 9) | 0x3f800000u) - 1.0f;
  return (u > 0.5f) ? 1.0f : 0.0f;
}

__device__ inline float gelu_exact(float x) {
  return 0.5f * x * (1.0f + erff(x * 0.70710678118654752f));
}

__device__ inline v8f splat8(float x) {
  v8f r;
  #pragma unroll
  for (int i = 0; i < 8; ++i) r[i] = x;
  return r;
}

// byte offset of patch element k (k = c*9 + (dy+1)*3 + (dx+1)) inside the
// [8][18][18] f32 halo tile, relative to (row=0, col=0) lane origin
__host__ __device__ constexpr int patch_off(int k) {
  return ((k / 9) * 324 + ((k % 9) / 3) * 18 + ((k % 9) % 3)) * 4;
}

// ---------------------------------------------------------------------------
// Bilinear upsample seed (1,8,8,8) -> (8,1024,1024), align_corners=False
// ---------------------------------------------------------------------------
__global__ void nca_upsample_kernel(const float *__restrict__ seed,
                                    float *__restrict__ x) {
  int idx = blockIdx.x * 256 + threadIdx.x;
  if (idx >= SDIM * HW) return;
  int c  = idx >> 20;
  int p  = idx & (HW - 1);
  int gy = p >> 10, gx = p & (IMG - 1);
  float sy = (gy + 0.5f) * (8.0f / IMG) - 0.5f;
  float sx = (gx + 0.5f) * (8.0f / IMG) - 0.5f;
  int y0 = (int)floorf(sy), x0 = (int)floorf(sx);
  float wy = sy - (float)y0, wx = sx - (float)x0;
  int y0c = y0 < 0 ? 0 : (y0 > 7 ? 7 : y0);
  int y1c = (y0 + 1) < 0 ? 0 : ((y0 + 1) > 7 ? 7 : (y0 + 1));
  int x0c = x0 < 0 ? 0 : (x0 > 7 ? 7 : x0);
  int x1c = (x0 + 1) < 0 ? 0 : ((x0 + 1) > 7 ? 7 : (x0 + 1));
  const float *s = seed + c * 64;
  float v00 = s[y0c * 8 + x0c], v01 = s[y0c * 8 + x1c];
  float v10 = s[y1c * 8 + x0c], v11 = s[y1c * 8 + x1c];
  x[idx] = v00 * (1.f - wy) * (1.f - wx) + v01 * (1.f - wy) * wx +
           v10 * wy * (1.f - wx) + v11 * wy * wx;
}

// ---------------------------------------------------------------------------
// One NCA step on a 16x16 pixel tile per workgroup (8 waves, wave32).
// ---------------------------------------------------------------------------
__global__ __launch_bounds__(256) void nca_step_kernel(
    const float *__restrict__ xin, float *__restrict__ xout,
    const float *__restrict__ W1, const float *__restrict__ b1,
    const float *__restrict__ W2, const float *__restrict__ b2,
    unsigned key0, unsigned key1) {
  // tilebuf = [8][18][18] halo'd x tile + 20-float zero pad (OOB-K target)
  __shared__ __align__(16) float    tilebuf[SDIM * 324 + 20];
  __shared__ __align__(16) float    w1pad[HID * 96];   // W1 zero-padded K 72->96
  __shared__ __align__(16) float    w2pad[16 * HID];   // W2 zero-padded N 8->16
  __shared__ __align__(16) float    masks[256];        // dropout mask per pixel
  __shared__ __align__(16) _Float16 hbuf[8][16][HID];  // per-wave hidden acts

  const int tid   = threadIdx.x;
  const int lane  = tid & 31;
  const int wid   = tid >> 5;
  const int baseY = blockIdx.y * 16;
  const int baseX = blockIdx.x * 16;
  const int ZB    = SDIM * 324 * 4;  // byte offset of the zero pad

#if USE_TDM
  const bool interior = (blockIdx.x != 0) && (blockIdx.x != (IMG / 16 - 1)) &&
                        (blockIdx.y != 0) && (blockIdx.y != (IMG / 16 - 1));
#else
  const bool interior = false;
#endif

  // ---- stage: weights (padded), zero pad, per-pixel threefry mask ----
  for (int idx = tid; idx < HID * 96; idx += 256) {
    int n = idx / 96, k = idx % 96;
    w1pad[idx] = (k < 72) ? W1[n * 72 + k] : 0.0f;
  }
  for (int idx = tid; idx < 16 * HID; idx += 256)
    w2pad[idx] = (idx < SDIM * HID) ? W2[idx] : 0.0f;
  if (tid < 20) tilebuf[SDIM * 324 + tid] = 0.0f;
  {
    int r = tid >> 4, cl = tid & 15;
    unsigned p = (unsigned)((baseY + r) * IMG + baseX + cl);
    masks[tid] = nca_mask(key0, key1, p);
  }

  // ---- stage: halo tile (TDM on interior tiles, checked loop on edges) ----
  if (!interior) {
    for (int idx = tid; idx < SDIM * 324; idx += 256) {
      int c = idx / 324, rem = idx % 324;
      int ly = rem / 18, lx = rem % 18;
      int gy = baseY + ly - 1, gx = baseX + lx - 1;
      float v = 0.0f;
      if (gy >= 0 && gy < IMG && gx >= 0 && gx < IMG)
        v = xin[c * HW + gy * IMG + gx];
      tilebuf[idx] = v;
    }
  }
#if USE_TDM
  else if (wid == 0) {
    // D# per CDNA5 ISA 8.3-8.6: 3-D tile 18x18x8, f32, strides IMG / HW
    unsigned long long ga =
        (unsigned long long)(uintptr_t)(const void *)
            (xin + (size_t)(baseY - 1) * IMG + (baseX - 1));
    unsigned ldsa = (unsigned)(uintptr_t)(void *)&tilebuf[0];
    u32x4 g0;
    g0[0] = 1u;                                    // count=1 (valid)
    g0[1] = ldsa;                                  // lds_addr
    g0[2] = (unsigned)(ga & 0xFFFFFFFFu);          // global_addr[31:0]
    g0[3] = (unsigned)((ga >> 32) & 0x01FFFFFFu) | (2u << 30);  // [56:32]|type=2
    i32x8 g1;
    g1[0] = (int)(2u << 16);                       // data_size=4B, mask=0
    g1[1] = (int)(18u << 16);                      // tensor_dim0[15:0]
    g1[2] = (int)(18u << 16);                      // tensor_dim1[15:0]
    g1[3] = (int)(18u << 16);                      // tile_dim0=18
    g1[4] = (int)(18u | (8u << 16));               // tile_dim1=18, tile_dim2=8
    g1[5] = (int)IMG;                              // tensor_dim0_stride
    g1[6] = 0;                                     // dim1_stride[15:0]=0 (HW lo)
    g1[7] = (int)(HW >> 16);                       // tensor_dim1_stride[47:16]
    i32x4 g2; g2[0] = SDIM; g2[1] = 0; g2[2] = 0; g2[3] = 0;  // tensor_dim2=8
    i32x4 g3; g3[0] = 0; g3[1] = 0; g3[2] = 0; g3[3] = 0;
#if __clang_major__ >= 23
    i32x8 gx; for (int i = 0; i < 8; ++i) gx[i] = 0;
    __builtin_amdgcn_tensor_load_to_lds(g0, g1, g2, g3, gx, 0);
#else
    __builtin_amdgcn_tensor_load_to_lds(g0, g1, g2, g3, 0);
#endif
    __builtin_amdgcn_s_wait_tensorcnt(0);
  }
#endif
  __syncthreads();

  const int nl     = lane & 15;
  const int hiHalf = (lane >= 16) ? 1 : 0;

  const float b1v0 = b1[nl];
  const float b1v1 = b1[nl + 16];
  const float b2v  = (nl < 8) ? b2[nl] : 0.0f;

  // ---- A-fragment LDS byte addresses, fully constant-folded per slot ----
  // slot (kb,e): K = 32*kb + e + 16*hiHalf ; OOB K -> zero pad
  const int laneA = wid * 144 + nl * 4;  // (wid*2 rows)*72B + col*4B
  int aoff[3][16];
  #pragma unroll
  for (int kb = 0; kb < 3; ++kb) {
    #pragma unroll
    for (int e = 0; e < 16; ++e) {
      const int kLo = 32 * kb + e;
      const int kHi = kLo + 16;
      const int oLo = (kLo < 72) ? (laneA + patch_off(kLo)) : ZB;
      const int oHi = (kHi < 72) ? (laneA + patch_off(kHi)) : ZB;
      aoff[kb][e] = hiHalf ? oHi : oLo;
    }
  }

  // ---- B1 fragments: [kstep][nchunk], K = e + 16*hiHalf + 32*kb ----
  const float4 *w1v = (const float4 *)w1pad;
  F16x16 B1[3][2];
  #pragma unroll
  for (int kb = 0; kb < 3; ++kb) {
    #pragma unroll
    for (int nc = 0; nc < 2; ++nc) {
      const int base = ((nc * 16 + nl) * 96 + kb * 32 + hiHalf * 16) >> 2;
      #pragma unroll
      for (int q = 0; q < 4; ++q) {
        float4 f = w1v[base + q];
        B1[kb][nc].h[q * 4 + 0] = (_Float16)f.x;
        B1[kb][nc].h[q * 4 + 1] = (_Float16)f.y;
        B1[kb][nc].h[q * 4 + 2] = (_Float16)f.z;
        B1[kb][nc].h[q * 4 + 3] = (_Float16)f.w;
      }
    }
  }
  // ---- B2 fragment: K = e + 16*hiHalf, column nl (zero for nl >= 8) ----
  const float4 *w2v = (const float4 *)w2pad;
  F16x16 B2;
  {
    const int base = (nl * HID + hiHalf * 16) >> 2;
    #pragma unroll
    for (int q = 0; q < 4; ++q) {
      float4 f = w2v[base + q];
      B2.h[q * 4 + 0] = (_Float16)f.x;
      B2.h[q * 4 + 1] = (_Float16)f.y;
      B2.h[q * 4 + 2] = (_Float16)f.z;
      B2.h[q * 4 + 3] = (_Float16)f.w;
    }
  }

  const char *tb = (const char *)tilebuf;

  // ---- each wave owns two pixel-rows of the tile ----
  #pragma unroll
  for (int bi = 0; bi < 2; ++bi) {
    const int row = wid * 2 + bi;  // 0..15
    v8f c0 = splat8(b1v0);
    v8f c1 = splat8(b1v1);
    #pragma unroll
    for (int kb = 0; kb < 3; ++kb) {
      F16x16 A;
      #pragma unroll
      for (int e = 0; e < 16; ++e)
        A.h[e] = (_Float16)*(const float *)(tb + aoff[kb][e] + bi * 72);
      c0 = __builtin_amdgcn_wmma_f32_16x16x32_f16(false, A.v, false,
                                                  B1[kb][0].v, (short)0, c0,
                                                  false, false);
      c1 = __builtin_amdgcn_wmma_f32_16x16x32_f16(false, A.v, false,
                                                  B1[kb][1].v, (short)0, c1,
                                                  false, false);
    }
    // GELU -> stash hidden to LDS (C layout: M = v + 8*hiHalf, N = nl)
    #pragma unroll
    for (int v = 0; v < 8; ++v) {
      int px = v + hiHalf * 8;
      hbuf[wid][px][nl]      = (_Float16)gelu_exact(c0[v]);
      hbuf[wid][px][nl + 16] = (_Float16)gelu_exact(c1[v]);
    }
    // same-wave LDS handoff: DS ops are in-order per wave; fence the compiler
    asm volatile("s_wait_dscnt 0" ::: "memory");

    // A2: 16 pixels x 32 hidden from hbuf
    F16x16 A2;
    #pragma unroll
    for (int e = 0; e < 16; ++e) {
      int hc = e + hiHalf * 16;
      A2.h[e] = hbuf[wid][nl][hc];
    }
    v8f c2 = splat8(b2v);
    c2 = __builtin_amdgcn_wmma_f32_16x16x32_f16(false, A2.v, false, B2.v,
                                                (short)0, c2, false, false);
    // x += 0.1 * dx * mask  (channel = nl, pixel col = v + 8*hiHalf)
    if (nl < 8) {
      #pragma unroll
      for (int v = 0; v < 8; ++v) {
        int px = v + hiHalf * 8;
        float dx = c2[v] * 0.1f;
        float m  = masks[row * 16 + px];
        float xi = tilebuf[nl * 324 + (row + 1) * 18 + (px + 1)];
        xout[nl * HW + (baseY + row) * IMG + baseX + px] = xi + dx * m;
      }
    }
  }
}

// ---------------------------------------------------------------------------
// Projection: out[h,w] = sum_c x[c,h,w] * Wp[c] + bp
// ---------------------------------------------------------------------------
__global__ void nca_project_kernel(const float *__restrict__ x,
                                   const float *__restrict__ Wp,
                                   const float *__restrict__ bp,
                                   float *__restrict__ out) {
  int p = blockIdx.x * 256 + threadIdx.x;
  if (p >= HW) return;
  float acc = bp[0];
  #pragma unroll
  for (int c = 0; c < SDIM; ++c) acc += x[c * HW + p] * Wp[c];
  out[p] = acc;
}

extern "C" void kernel_launch(void *const *d_in, const int *in_sizes, int n_in,
                              void *d_out, int out_size, void *d_ws,
                              size_t ws_size, hipStream_t stream) {
  (void)in_sizes; (void)n_in; (void)out_size; (void)ws_size;
  const float *seed = (const float *)d_in[0];
  const float *W1   = (const float *)d_in[1];
  const float *b1   = (const float *)d_in[2];
  const float *W2   = (const float *)d_in[3];
  const float *b2   = (const float *)d_in[4];
  const float *Wp   = (const float *)d_in[5];
  const float *bp   = (const float *)d_in[6];
  float *out = (float *)d_out;

  // ping-pong state buffers: 2 x 32 MB in workspace
  float *bufA = (float *)d_ws;
  float *bufB = bufA + (size_t)SDIM * HW;

  // host-side jax.random.split(jax.random.key(42), 64)
  unsigned word[128];
  for (int j = 0; j < 64; ++j) {
    unsigned o0, o1;
    threefry2x32(0u, 42u, (unsigned)j, (unsigned)(j + 64), o0, o1);
    word[j] = o0; word[64 + j] = o1;
  }

  nca_upsample_kernel<<<(SDIM * HW + 255) / 256, 256, 0, stream>>>(seed, bufA);

  float *cur = bufA, *nxt = bufB;
  dim3 grid(IMG / 16, IMG / 16);
  for (int it = 0; it < 64; ++it) {
    nca_step_kernel<<<grid, 256, 0, stream>>>(cur, nxt, W1, b1, W2, b2,
                                              word[2 * it], word[2 * it + 1]);
    float *t = cur; cur = nxt; nxt = t;
  }

  nca_project_kernel<<<(HW + 255) / 256, 256, 0, stream>>>(cur, Wp, bp, out);
}
